// slot_attn_76622216561031
// MI455X (gfx1250) — matrix operand order repaired and checked by reading, not compiled, not run on hardware
//
#include <hip/hip_runtime.h>
#include <hip/hip_bf16.h>

// ---------------- problem constants ----------------
enum : int {
  BB = 64, NN = 4096, KK = 16, DD = 256, HID = 512, TT = 3,
  TN = 32,      // inner n-tile rows
  TNB = 128,    // rows per block
  NCHUNK = NN / TNB
};

typedef __attribute__((ext_vector_type(16))) __bf16 v16bf;
typedef __attribute__((ext_vector_type(2)))  __bf16 bf16x2;
typedef __attribute__((ext_vector_type(8)))  float  v8f;

// ---------------- optional CDNA5 async global->LDS path ----------------
#if __has_builtin(__builtin_amdgcn_global_load_async_to_lds_b128) && \
    __has_builtin(__builtin_amdgcn_s_wait_asynccnt)
#define HAVE_ASYNC_LDS 1
#else
#define HAVE_ASYNC_LDS 0
#endif

#if HAVE_ASYNC_LDS
#define AS1 __attribute__((address_space(1)))
#define AS3 __attribute__((address_space(3)))
typedef int v4i_raw __attribute__((vector_size(16)));
__device__ __forceinline__ void async_copy_b128(const void* g, void* l) {
  __builtin_amdgcn_global_load_async_to_lds_b128((AS1 v4i_raw*)g,
                                                 (AS3 v4i_raw*)l, 0, 0);
}
#endif

// ---------------- WMMA helpers (wave32, 16x16x32 bf16) ----------------
__device__ __forceinline__ v8f wmma_bf16(v16bf a, v16bf b, v8f c) {
  return __builtin_amdgcn_wmma_f32_16x16x32_bf16(false, a, false, b, (short)0, c,
                                                 false, false);
}

// A fragment 16x32 (MxK), row-major source, row stride ld (elements),
// src pre-offset to (row 0, k0). Lanes rows M=0..15 hold K = h*16+0..15,
// h = lane/16. Contiguous 16-element run.
__device__ __forceinline__ v16bf load_a_frag(const __bf16* src, int ld, int lane) {
  int m = lane & 15, h = lane >> 4;
  const __bf16* p0 = src + m * ld + h * 16;
  const __bf16* p1 = p0 + 8;
  v16bf a;
#pragma unroll
  for (int j = 0; j < 8; ++j) { a[j] = p0[j]; a[8 + j] = p1[j]; }
  return a;
}

// B fragment 32x16 (KxN) where the logical B[k][n] = W[n][k] (W row-major
// [n][k], row stride ldw). src pre-offset to (n0, k0). Lane reads a
// CONTIGUOUS run of 16 elements along k -> 2x ds/global b128.
__device__ __forceinline__ v16bf load_bT_frag(const __bf16* w, int ldw, int lane) {
  int n = lane & 15, h = lane >> 4;
  const __bf16* p = w + n * ldw + h * 16;
  v16bf b;
#pragma unroll
  for (int e = 0; e < 16; ++e) b[e] = p[e];
  return b;
}

// ---------------- block reduction (sum, sumsq) ----------------
template <int NT>
__device__ __forceinline__ void block_reduce2(float& s1, float& s2, float* sm) {
  int lane = threadIdx.x & 31, wv = threadIdx.x >> 5;
#pragma unroll
  for (int off = 16; off > 0; off >>= 1) {
    s1 += __shfl_down(s1, off, 32);
    s2 += __shfl_down(s2, off, 32);
  }
  if (lane == 0) { sm[wv] = s1; sm[NT / 32 + wv] = s2; }
  __syncthreads();
  if (wv == 0) {
    float a = (lane < NT / 32) ? sm[lane] : 0.f;
    float b = (lane < NT / 32) ? sm[NT / 32 + lane] : 0.f;
#pragma unroll
    for (int off = 16; off > 0; off >>= 1) {
      a += __shfl_down(a, off, 32);
      b += __shfl_down(b, off, 32);
    }
    if (lane == 0) { sm[0] = a; sm[1] = b; }
  }
  __syncthreads();
  s1 = sm[0]; s2 = sm[1];
}

// ---------------- prep kernels ----------------
__global__ void f32_to_bf16_kernel(const float* __restrict__ src,
                                   __bf16* __restrict__ dst, int n) {
  int i = blockIdx.x * blockDim.x + threadIdx.x;
  if (i < n) dst[i] = (__bf16)src[i];
}

__global__ void init_slots_kernel(const float* __restrict__ si,
                                  const float* __restrict__ mu,
                                  const float* __restrict__ ls,
                                  float* __restrict__ slots, int n) {
  int i = blockIdx.x * blockDim.x + threadIdx.x;
  if (i < n) {
    int d = i & (DD - 1);
    slots[i] = mu[d] + __expf(ls[d]) * si[i];
  }
}

// LN over (N,DIN) per batch, write bf16 x
__global__ __launch_bounds__(1024) void ln0_kernel(const float* __restrict__ x,
                                                   const float* __restrict__ w,
                                                   const float* __restrict__ bia,
                                                   __bf16* __restrict__ out) {
  __shared__ float sm[64];
  int b = blockIdx.x;
  const float* xb = x + (size_t)b * NN * DD;
  float s1 = 0.f, s2 = 0.f;
  for (int i = threadIdx.x; i < NN * DD; i += 1024) {
    float v = xb[i];
    s1 += v; s2 += v * v;
  }
  block_reduce2<1024>(s1, s2, sm);
  const float inv = 1.0f / (float)(NN * DD);
  float m = s1 * inv;
  float rs = rsqrtf(s2 * inv - m * m + 1e-5f);
  __bf16* ob = out + (size_t)b * NN * DD;
  for (int i = threadIdx.x; i < NN * DD; i += 1024)
    ob[i] = (__bf16)((xb[i] - m) * rs * w[i] + bia[i]);
}

// ---------------- per-iteration: LN1(slots) -> q (WMMA), zero U/S ----------------
__global__ __launch_bounds__(512) void slots_q_kernel(
    const float* __restrict__ slots, const float* __restrict__ w1,
    const float* __restrict__ b1, const __bf16* __restrict__ qWb,
    const float* __restrict__ qb, __bf16* __restrict__ qout,
    float* __restrict__ U, float* __restrict__ S) {
  __shared__ __bf16 lds_s[16 * 264];
  __shared__ float sm[64];
  int b = blockIdx.x;
  const float* sb = slots + (size_t)b * (KK * DD);
  float vals[8];
  float s1 = 0.f, s2 = 0.f;
#pragma unroll
  for (int i = 0; i < 8; ++i) {
    float v = sb[threadIdx.x + 512 * i];
    vals[i] = v; s1 += v; s2 += v * v;
  }
  block_reduce2<512>(s1, s2, sm);
  const float inv = 1.0f / (float)(KK * DD);
  float m = s1 * inv;
  float rs = rsqrtf(s2 * inv - m * m + 1e-5f);
#pragma unroll
  for (int i = 0; i < 8; ++i) {
    int idx = threadIdx.x + 512 * i;
    lds_s[(idx >> 8) * 264 + (idx & 255)] =
        (__bf16)((vals[i] - m) * rs * w1[idx] + b1[idx]);
  }
  // zero accumulators for this iteration
#pragma unroll
  for (int i = 0; i < 8; ++i) U[(size_t)b * (KK * DD) + threadIdx.x + 512 * i] = 0.f;
  if (threadIdx.x < KK) S[b * KK + threadIdx.x] = 0.f;
  __syncthreads();

  int lane = threadIdx.x & 31, wv = threadIdx.x >> 5;
  int e0 = wv * 16;
  v8f acc = {};
#pragma unroll
  for (int ks = 0; ks < 8; ++ks) {
    v16bf a = load_a_frag(lds_s + ks * 32, 264, lane);
    v16bf bb = load_bT_frag(qWb + e0 * DD + ks * 32, DD, lane);
    acc = wmma_bf16(a, bb, acc);
  }
  int nl = lane & 15, hf = lane >> 4;
  float bias = qb[e0 + nl];
#pragma unroll
  for (int r = 0; r < 8; ++r)
    qout[(size_t)b * (KK * DD) + (r + 8 * hf) * DD + e0 + nl] = (__bf16)(acc[r] + bias);
}

// ---------------- the heavy fused streaming kernel ----------------
// Per (n-chunk, batch): k = x@kW^T+kb (stored natural [n][e]) -> dist =
// q@k^T/16 (WMMA, B-frags contiguous) -> column softmax over the 16 slots ->
// v = x@vW^T+vb (stored transposed [d][n]) -> U += attn@v, S += rowsum(attn).
// x tiles are double-buffered: async DMA of tile t+1 overlaps compute of t.
__global__ __launch_bounds__(256) void attn_stream_kernel(
    const __bf16* __restrict__ xbf, const __bf16* __restrict__ qbf,
    const __bf16* __restrict__ kWb, const float* __restrict__ kb,
    const __bf16* __restrict__ vWb, const float* __restrict__ vb,
    float* __restrict__ U, float* __restrict__ S) {
  __shared__ __bf16 lds_x[2][TN * 264];   // ping-pong x tiles, row-major [n][d]
  __shared__ __bf16 lds_kv[256 * 40];     // k natural [n][e] (stride 264), then vT [d][n] (stride 40)
  __shared__ __bf16 lds_q[16 * 264];      // q for this batch
  __shared__ __bf16 lds_attn[16 * 40];    // attn [slot][n], A-source for U update
  __shared__ float lds_S[KK];

  int b = blockIdx.y;
  int n_base = blockIdx.x * TNB;
  int lane = threadIdx.x & 31, wv = threadIdx.x >> 5;
  int nl = lane & 15, hf = lane >> 4;

  const __bf16* xchunk = xbf + ((size_t)b * NN + n_base) * DD;

#if HAVE_ASYNC_LDS
  // kick off the DMA for tile 0; it overlaps the q preload below
  for (int j = threadIdx.x; j < TN * 32; j += 256) {
    int rr = j >> 5, c16 = (j & 31) * 8;
    async_copy_b128(xchunk + rr * DD + c16, &lds_x[0][rr * 264 + c16]);
  }
#endif

  for (int i = threadIdx.x; i < 16 * DD; i += 256) {
    int rr = i >> 8, cc = i & 255;
    lds_q[rr * 264 + cc] = qbf[(size_t)b * (KK * DD) + i];
  }
  if (threadIdx.x < KK) lds_S[threadIdx.x] = 0.f;
  v8f uacc0 = {}, uacc1 = {};  // wave owns d-tiles 2*wv, 2*wv+1

#if HAVE_ASYNC_LDS
  __builtin_amdgcn_s_wait_asynccnt(0);
#endif
  __syncthreads();

  int cur = 0;
  for (int t0 = 0; t0 < TNB; t0 += TN) {
#if HAVE_ASYNC_LDS
    // issue DMA for the NEXT tile into the other buffer; no wait here --
    // it runs in the background across this whole iteration's compute.
    if (t0 + TN < TNB) {
      const __bf16* xnext = xchunk + (size_t)(t0 + TN) * DD;
      for (int j = threadIdx.x; j < TN * 32; j += 256) {
        int rr = j >> 5, c16 = (j & 31) * 8;
        async_copy_b128(xnext + rr * DD + c16, &lds_x[cur ^ 1][rr * 264 + c16]);
      }
    }
#else
    {
      const __bf16* xrow = xchunk + (size_t)t0 * DD;
      for (int i = threadIdx.x; i < TN * DD; i += 256) {
        int rr = i >> 8, cc = i & 255;
        lds_x[cur][rr * 264 + cc] = xrow[i];
      }
      __builtin_prefetch(xrow + (size_t)TN * DD, 0, 0);
      __syncthreads();
    }
#endif
    const __bf16* xcur = lds_x[cur];

    // ---- K projection, natural store: k[n][e] (stride 264) ----
#pragma unroll
    for (int tt = 0; tt < 4; ++tt) {
      int tile = wv * 4 + tt;
      int rt = tile & 1;           // n row-tile
      int e0 = (tile >> 1) * 16;   // e col-tile
      v8f c = {};
#pragma unroll
      for (int ks = 0; ks < 8; ++ks) {
        v16bf a = load_a_frag(xcur + rt * 16 * 264 + ks * 32, 264, lane);
        v16bf bb = load_bT_frag(kWb + e0 * DD + ks * 32, DD, lane);
        c = wmma_bf16(a, bb, c);
      }
      float bias = kb[e0 + nl];
#pragma unroll
      for (int r = 0; r < 8; ++r) {
        int nloc = rt * 16 + r + 8 * hf;
        lds_kv[nloc * 264 + e0 + nl] = (__bf16)(c[r] + bias);
      }
    }
    __syncthreads();

    // ---- dist = q @ k^T / 16, softmax over slot dim (per column) ----
    // B[k=e][col=n] = k[n][e] -> contiguous bT fragment reads from lds_kv.
    if (wv < 2) {
      int nt = wv;
      v8f c = {};
#pragma unroll
      for (int ks = 0; ks < 8; ++ks) {
        v16bf a = load_a_frag(lds_q + ks * 32, 264, lane);
        v16bf bb = load_bT_frag(lds_kv + (nt * 16) * 264 + ks * 32, 264, lane);
        c = wmma_bf16(a, bb, c);
      }
      float mx = -1e30f;
#pragma unroll
      for (int r = 0; r < 8; ++r) { c[r] *= 0.0625f; mx = fmaxf(mx, c[r]); }
      mx = fmaxf(mx, __shfl_xor(mx, 16, 32));
      float sum = 0.f;
#pragma unroll
      for (int r = 0; r < 8; ++r) { c[r] = __expf(c[r] - mx); sum += c[r]; }
      sum += __shfl_xor(sum, 16, 32);
      float rinv = 1.0f / sum;
#pragma unroll
      for (int r = 0; r < 8; ++r) {
        float av = c[r] * rinv;
        int slot = r + 8 * hf;
        lds_attn[slot * 40 + nt * 16 + nl] = (__bf16)av;
        atomicAdd(&lds_S[slot], av);
      }
    }
    __syncthreads();

    // ---- V projection, transposed store: vT[d][n] (stride 40), packed b32 ----
#pragma unroll
    for (int tt = 0; tt < 4; ++tt) {
      int tile = wv * 4 + tt;
      int rt = tile & 1;
      int d0 = (tile >> 1) * 16;
      v8f c = {};
#pragma unroll
      for (int ks = 0; ks < 8; ++ks) {
        v16bf a = load_a_frag(xcur + rt * 16 * 264 + ks * 32, 264, lane);
        v16bf bb = load_bT_frag(vWb + d0 * DD + ks * 32, DD, lane);
        c = wmma_bf16(a, bb, c);
      }
      float bias = vb[d0 + nl];
      int nbase = rt * 16 + 8 * hf;
#pragma unroll
      for (int r = 0; r < 8; r += 2) {
        bf16x2 pk = { (__bf16)(c[r] + bias), (__bf16)(c[r + 1] + bias) };
        *(bf16x2*)&lds_kv[(d0 + nl) * 40 + nbase + r] = pk;
      }
    }
    __syncthreads();

    // ---- U += attn @ v  (K = TN = 32, one WMMA per d-tile) ----
    // B[k=n][col=d] = vT[d][n] -> contiguous bT fragment reads.
    {
      v16bf a = load_a_frag(lds_attn, 40, lane);
      v16bf b0 = load_bT_frag(lds_kv + (2 * wv) * 16 * 40, 40, lane);
      uacc0 = wmma_bf16(a, b0, uacc0);
      v16bf b1 = load_bT_frag(lds_kv + (2 * wv + 1) * 16 * 40, 40, lane);
      uacc1 = wmma_bf16(a, b1, uacc1);
    }
#if HAVE_ASYNC_LDS
    __builtin_amdgcn_s_wait_asynccnt(0);  // next tile's DMA must be in LDS
#endif
    __syncthreads();
    cur ^= 1;
  }

  // flush partials
#pragma unroll
  for (int r = 0; r < 8; ++r) {
    int slot = r + 8 * hf;
    atomicAdd(&U[((size_t)b * KK + slot) * DD + 2 * wv * 16 + nl], uacc0[r]);
    atomicAdd(&U[((size_t)b * KK + slot) * DD + (2 * wv + 1) * 16 + nl], uacc1[r]);
  }
  if (threadIdx.x < KK) atomicAdd(&S[b * KK + threadIdx.x], lds_S[threadIdx.x]);
}

// ---------------- finish: slots = U/(S+eps); LN2 + MLP (WMMA) + residual ----------------
__global__ __launch_bounds__(512) void slots_finish_kernel(
    const float* __restrict__ U, const float* __restrict__ S,
    const float* __restrict__ w2, const float* __restrict__ b2,
    const __bf16* __restrict__ m1Wb, const float* __restrict__ m1b,
    const __bf16* __restrict__ m2Wb, const float* __restrict__ m2b,
    float* __restrict__ dst) {
  __shared__ float lds_sn[16 * 264];
  __shared__ __bf16 lds_h[16 * 264];
  __shared__ __bf16 lds_h1[16 * 520];
  __shared__ float sm[64];
  int b = blockIdx.x;
  float vals[8];
  float s1 = 0.f, s2 = 0.f;
#pragma unroll
  for (int i = 0; i < 8; ++i) {
    int idx = threadIdx.x + 512 * i;
    int slot = idx >> 8;
    float v = U[(size_t)b * (KK * DD) + idx] / (S[b * KK + slot] + 1e-7f);
    vals[i] = v; s1 += v; s2 += v * v;
    lds_sn[slot * 264 + (idx & 255)] = v;
  }
  block_reduce2<512>(s1, s2, sm);
  const float inv = 1.0f / (float)(KK * DD);
  float m = s1 * inv;
  float rs = rsqrtf(s2 * inv - m * m + 1e-5f);
#pragma unroll
  for (int i = 0; i < 8; ++i) {
    int idx = threadIdx.x + 512 * i;
    lds_h[(idx >> 8) * 264 + (idx & 255)] =
        (__bf16)((vals[i] - m) * rs * w2[idx] + b2[idx]);
  }
  __syncthreads();

  int lane = threadIdx.x & 31, wv = threadIdx.x >> 5;
  int nl = lane & 15, hf = lane >> 4;
  // h1 = relu(h @ m1W^T + m1b) : 32 col tiles over HID, 2 per wave
#pragma unroll
  for (int t = 0; t < 2; ++t) {
    int h0 = (wv + t * 16) * 16;
    v8f c = {};
#pragma unroll
    for (int ks = 0; ks < 8; ++ks) {
      v16bf a = load_a_frag(lds_h + ks * 32, 264, lane);
      v16bf bb = load_bT_frag(m1Wb + h0 * DD + ks * 32, DD, lane);
      c = wmma_bf16(a, bb, c);
    }
    float bias = m1b[h0 + nl];
#pragma unroll
    for (int r = 0; r < 8; ++r) {
      float v = c[r] + bias;
      v = v > 0.f ? v : 0.f;
      lds_h1[(r + 8 * hf) * 520 + h0 + nl] = (__bf16)v;
    }
  }
  __syncthreads();
  // out = snew + h1 @ m2W^T + m2b : one d-tile per wave, K = HID = 512
  {
    int d0 = wv * 16;
    v8f c = {};
#pragma unroll
    for (int ks = 0; ks < 16; ++ks) {
      v16bf a = load_a_frag(lds_h1 + ks * 32, 520, lane);
      v16bf bb = load_bT_frag(m2Wb + d0 * HID + ks * 32, HID, lane);
      c = wmma_bf16(a, bb, c);
    }
    float bias = m2b[d0 + nl];
#pragma unroll
    for (int r = 0; r < 8; ++r) {
      int slot = r + 8 * hf;
      dst[(size_t)b * (KK * DD) + slot * DD + d0 + nl] =
          c[r] + bias + lds_sn[slot * 264 + d0 + nl];
    }
  }
}

// ---------------- host launcher ----------------
extern "C" void kernel_launch(void* const* d_in, const int* in_sizes, int n_in,
                              void* d_out, int out_size, void* d_ws, size_t ws_size,
                              hipStream_t stream) {
  (void)in_sizes; (void)n_in; (void)out_size; (void)ws_size;
  const float* inputs     = (const float*)d_in[0];
  const float* slots_init = (const float*)d_in[1];
  const float* mu         = (const float*)d_in[2];
  const float* logsigma   = (const float*)d_in[3];
  const float* ln0_w      = (const float*)d_in[4];
  const float* ln0_b      = (const float*)d_in[5];
  const float* ln1_w      = (const float*)d_in[6];
  const float* ln1_b      = (const float*)d_in[7];
  const float* ln2_w      = (const float*)d_in[8];
  const float* ln2_b      = (const float*)d_in[9];
  const float* qW         = (const float*)d_in[10];
  const float* qb         = (const float*)d_in[11];
  const float* kW         = (const float*)d_in[12];
  const float* kb         = (const float*)d_in[13];
  const float* vW         = (const float*)d_in[14];
  const float* vb         = (const float*)d_in[15];
  const float* m1W        = (const float*)d_in[16];
  const float* m1b        = (const float*)d_in[17];
  const float* m2W        = (const float*)d_in[18];
  const float* m2b        = (const float*)d_in[19];

  char* p = (char*)d_ws;
  auto carve = [&](size_t bytes) -> char* {
    char* r = p;
    p += (bytes + 255) & ~(size_t)255;
    return r;
  };
  __bf16* x_bf   = (__bf16*)carve((size_t)BB * NN * DD * 2);
  __bf16* q_bf   = (__bf16*)carve((size_t)BB * KK * DD * 2);
  float*  slots  = (float*) carve((size_t)BB * KK * DD * 4);
  float*  U      = (float*) carve((size_t)BB * KK * DD * 4);
  float*  S      = (float*) carve((size_t)BB * KK * 4);
  __bf16* qW_bf  = (__bf16*)carve((size_t)TT * DD * DD * 2);
  __bf16* kW_bf  = (__bf16*)carve((size_t)TT * DD * DD * 2);
  __bf16* vW_bf  = (__bf16*)carve((size_t)TT * DD * DD * 2);
  __bf16* m1W_bf = (__bf16*)carve((size_t)TT * HID * DD * 2);
  __bf16* m2W_bf = (__bf16*)carve((size_t)TT * DD * HID * 2);

  const int nW = TT * DD * DD;       // 196608
  const int nM = TT * HID * DD;      // 393216
  f32_to_bf16_kernel<<<(nW + 255) / 256, 256, 0, stream>>>(qW, qW_bf, nW);
  f32_to_bf16_kernel<<<(nW + 255) / 256, 256, 0, stream>>>(kW, kW_bf, nW);
  f32_to_bf16_kernel<<<(nW + 255) / 256, 256, 0, stream>>>(vW, vW_bf, nW);
  f32_to_bf16_kernel<<<(nM + 255) / 256, 256, 0, stream>>>(m1W, m1W_bf, nM);
  f32_to_bf16_kernel<<<(nM + 255) / 256, 256, 0, stream>>>(m2W, m2W_bf, nM);

  const int nS = BB * KK * DD;
  init_slots_kernel<<<(nS + 255) / 256, 256, 0, stream>>>(slots_init, mu, logsigma,
                                                          slots, nS);
  ln0_kernel<<<BB, 1024, 0, stream>>>(inputs, ln0_w, ln0_b, x_bf);

  for (int it = 0; it < TT; ++it) {
    slots_q_kernel<<<BB, 512, 0, stream>>>(slots, ln1_w, ln1_b,
                                           qW_bf + (size_t)it * DD * DD,
                                           qb + it * DD, q_bf, U, S);
    attn_stream_kernel<<<dim3(NCHUNK, BB), 256, 0, stream>>>(
        x_bf, q_bf, kW_bf + (size_t)it * DD * DD, kb + it * DD,
        vW_bf + (size_t)it * DD * DD, vb + it * DD, U, S);
    float* dst = (it == TT - 1) ? (float*)d_out : slots;
    slots_finish_kernel<<<BB, 512, 0, stream>>>(
        U, S, ln2_w, ln2_b, m1W_bf + (size_t)it * HID * DD, m1b + it * HID,
        m2W_bf + (size_t)it * DD * HID, m2b + it * DD, dst);
  }
}
